// WindowAttention_42717744726498
// MI455X (gfx1250) — compile-verified, matrix operand order checked by reading, not run on hardware
//
#include <hip/hip_runtime.h>
#include <hip/hip_bf16.h>

// WindowAttention for MI455X (gfx1250), wave32 + WMMA bf16.
// B=4096 windows, N=64 tokens, C=256, H=8 heads, hd=32, top-19 masking.

#define WA_N    64    // tokens per window
#define WA_C    256   // channels
#define WA_H    8     // heads
#define WA_HD   32    // head dim
#define WA_QKVC 768   // 3*C
#define WA_TOPK 19    // int(64 * 0.3)

typedef __attribute__((ext_vector_type(16))) __bf16 v16bf;
typedef __attribute__((ext_vector_type(8)))  __bf16 v8bf;
typedef __attribute__((ext_vector_type(8)))  float  v8f;

static __device__ __forceinline__ v8f wmma_bf16(v16bf a, v16bf b, v8f c) {
  // D = A(16x32 bf16) * B(32x16 bf16) + C(16x16 f32)
  return __builtin_amdgcn_wmma_f32_16x16x32_bf16(false, a, false, b, (short)0, c,
                                                 false, false);
}

// Fragment lane layout (ISA 7.12.2, 16-bit operand, 16x32 row-major tile):
// lanes 0-15: row=lane, cols {0..7,16..23}; lanes 16-31: row=lane-16, cols {8..15,24..31}.

// bf16 tile (LDS or global) -> fragment via two 16B vector loads per lane.
static __device__ __forceinline__ v16bf frag_from_bf16(const __bf16* base, int stride,
                                                       int lane) {
  const int r  = lane & 15;
  const int cb = (lane >> 4) << 3;   // 0 or 8
  const __bf16* p = base + (size_t)r * stride + cb;
  const v8bf lo = *(const v8bf*)p;
  const v8bf hi = *(const v8bf*)(p + 16);
  v16bf f;
#pragma unroll
  for (int j = 0; j < 8; ++j) { f[j] = lo[j]; f[j + 8] = hi[j]; }
  return f;
}

// fp32 row-major tile -> bf16 fragment (fallback when no ws for weights).
static __device__ __forceinline__ v16bf frag_from_f32(const float* __restrict__ base,
                                                      int stride, int lane) {
  const int r  = lane & 15;
  const int cb = (lane >> 4) << 3;
  const float* p = base + (size_t)r * stride + cb;
  v16bf f;
#pragma unroll
  for (int j = 0; j < 8; ++j) {
    f[j]     = (__bf16)p[j];
    f[j + 8] = (__bf16)p[j + 16];
  }
  return f;
}

// ---------------------------------------------------------------------------
// Prologue: one-time f32 -> bf16 conversion of both weight matrices into ws.
// ---------------------------------------------------------------------------
__global__ __launch_bounds__(256)
void wa_cvt_w(const float* __restrict__ qw, const float* __restrict__ pw,
              __bf16* __restrict__ qwb, __bf16* __restrict__ pwb) {
  const int i = blockIdx.x * 256 + threadIdx.x;
  if (i < WA_QKVC * WA_C) qwb[i] = (__bf16)qw[i];
  if (i < WA_C * WA_C)    pwb[i] = (__bf16)pw[i];
}

// ---------------------------------------------------------------------------
// Kernel A: fused QKV GEMM + LayerNorm(q,k) + QK^T + top-k softmax + PV.
// One block per window; 8 waves, wave h owns head h for the attention phases.
// WB: weights preconverted to bf16 in ws.  OB: attention output as bf16 in ws.
// ---------------------------------------------------------------------------
template <bool WB, bool OB>
__global__ __launch_bounds__(256)
void wa_qkv_attn(const float* __restrict__ x,        // [B,64,256]
                 const float* __restrict__ qkv_w,    // [768,256] f32
                 const __bf16* __restrict__ qwb,     // [768,256] bf16 (WB)
                 const float* __restrict__ qkv_b,    // [768]
                 const float* __restrict__ q_gamma, const float* __restrict__ q_beta,
                 const float* __restrict__ k_gamma, const float* __restrict__ k_beta,
                 float*  __restrict__ aout_f,        // [B,64,256] f32  (!OB)
                 __bf16* __restrict__ aout_b)        // [B,64,256] bf16 (OB)
{
  // Region 1 (128KB): qkv staging -> scores S -> output staging.
  __shared__ __align__(16) union R1 {
    __bf16 qkv[WA_N][WA_QKVC];           // phase 1-2   (98304 B)
    float  S[WA_H][WA_N][WA_N];          // phase 3-4   (131072 B)
    float  ostg_f[WA_N][WA_C];           // phase 5-6   (65536 B)
    __bf16 ostg_b[WA_N][WA_C];           // phase 5-6   (32768 B)
  } r1;
  // Region 2 (64KB): LN'd q (pre-scaled) + k, then probabilities P.
  __shared__ __align__(16) union R2 {
    struct QK {
      __bf16 qs [WA_H][WA_N][WA_HD];
      __bf16 kln[WA_H][WA_N][WA_HD];
    } qk;
    __bf16 P[WA_H][WA_N][WA_N];
  } r2;
  // Region 3 (32KB): bf16 x tile, then v transposed.
  __shared__ __align__(16) union R3 {
    __bf16 xt[WA_N][WA_C];
    __bf16 vt[WA_H][WA_HD][WA_N];
  } r3;

  const int b    = blockIdx.x;
  const int tid  = threadIdx.x;
  const int lane = tid & 31;
  const int wave = tid >> 5;
  const float* xb = x + (size_t)b * WA_N * WA_C;

  // ---- Phase 0: stage x tile as bf16 (one conversion per element) ---------
  for (int e = tid; e < WA_N * WA_C; e += 256)
    r3.xt[e >> 8][e & 255] = (__bf16)xb[e];
  __syncthreads();

  // ---- Phase 1: qkv = x @ qkv_w^T + qkv_b  (64x768, tiles 4 x 48) ----------
  for (int t = wave; t < 4 * 48; t += 8) {
    const int mt = t & 3;        // token tile
    const int ct = t >> 2;       // output-column tile (0..47)
    v8f acc = {};
#pragma unroll
    for (int kk = 0; kk < WA_C; kk += 32) {
      v16bf a  = frag_from_bf16(&r3.xt[mt * 16][kk], WA_C, lane);
      v16bf wf = WB ? frag_from_bf16(qwb + (size_t)(ct * 16) * WA_C + kk, WA_C, lane)
                    : frag_from_f32(qkv_w + (size_t)(ct * 16) * WA_C + kk, WA_C, lane);
      acc = wmma_bf16(a, wf, acc);
    }
    const int   col   = ct * 16 + (lane & 15);
    const float bias  = qkv_b[col];
    const int   rbase = mt * 16 + ((lane >> 4) << 3);
#pragma unroll
    for (int v = 0; v < 8; ++v)
      r1.qkv[rbase + v][col] = (__bf16)(acc[v] + bias);
  }
  __syncthreads();

  // ---- Phase 2: LayerNorm q,k over hd=32 (+ fold in hd^-0.5 on q); V^T -----
  const float qscale = 0.17677669529663687f;  // 32^-0.5
  for (int task = tid; task < 2 * WA_H * WA_N; task += 256) {
    const int which = task >> 9;        // 0 = q, 1 = k
    const int h     = (task >> 6) & 7;
    const int n     = task & 63;
    const __bf16* src = &r1.qkv[n][which * WA_C + h * WA_HD];
    float vals[WA_HD];
    float mu = 0.f;
#pragma unroll
    for (int d = 0; d < WA_HD; ++d) { vals[d] = (float)src[d]; mu += vals[d]; }
    mu *= (1.f / WA_HD);
    float var = 0.f;
#pragma unroll
    for (int d = 0; d < WA_HD; ++d) { float dd = vals[d] - mu; var += dd * dd; }
    var *= (1.f / WA_HD);
    const float inv = rsqrtf(var + 1e-5f);
    if (which == 0) {
#pragma unroll
      for (int d = 0; d < WA_HD; ++d)
        r2.qk.qs[h][n][d] =
            (__bf16)(((vals[d] - mu) * inv * q_gamma[d] + q_beta[d]) * qscale);
    } else {
#pragma unroll
      for (int d = 0; d < WA_HD; ++d)
        r2.qk.kln[h][n][d] = (__bf16)((vals[d] - mu) * inv * k_gamma[d] + k_beta[d]);
    }
  }
  // v transpose: vt[h][d][n] = v[h][n][d]  (overwrites dead xt)
  for (int e = tid; e < WA_H * WA_HD * WA_N; e += 256) {
    const int h = e >> 11;
    const int d = (e >> 6) & 31;
    const int n = e & 63;
    r3.vt[h][d][n] = r1.qkv[n][2 * WA_C + h * WA_HD + d];
  }
  __syncthreads();

  // ---- Phase 3: S = (q*scale) @ k^T per head (hd=32 -> 1 WMMA per tile) ----
  const int h = wave;  // wave h owns head h
#pragma unroll
  for (int mt = 0; mt < 4; ++mt) {
    v16bf a = frag_from_bf16(&r2.qk.qs[h][mt * 16][0], WA_HD, lane);
#pragma unroll
    for (int nt = 0; nt < 4; ++nt) {
      v16bf kf = frag_from_bf16(&r2.qk.kln[h][nt * 16][0], WA_HD, lane);
      v8f acc = {};
      acc = wmma_bf16(a, kf, acc);
      const int col   = nt * 16 + (lane & 15);
      const int rbase = mt * 16 + ((lane >> 4) << 3);
#pragma unroll
      for (int v = 0; v < 8; ++v)
        r1.S[h][rbase + v][col] = acc[v];
    }
  }
  __syncthreads();  // P (below) overwrites qs/kln which other waves read above

  // ---- Phase 4: per-row top-19 + softmax (wave-cooperative) ----------------
  for (int row = 0; row < WA_N; ++row) {
    const float s0 = r1.S[h][row][lane];
    const float s1 = r1.S[h][row][lane + 32];
    bool  k0 = false, k1 = false;
    float rowmax = -3.4e38f;
#pragma unroll 1
    for (int t = 0; t < WA_TOPK; ++t) {
      const float c0 = k0 ? -3.4e38f : s0;
      const float c1 = k1 ? -3.4e38f : s1;
      float cv;
      int   ci;
      if (c1 > c0) { cv = c1; ci = lane + 32; } else { cv = c0; ci = lane; }
#pragma unroll
      for (int off = 16; off > 0; off >>= 1) {  // argmax, low-index tie-break
        const float ov = __shfl_xor(cv, off, 32);
        const int   oi = __shfl_xor(ci, off, 32);
        if (ov > cv || (ov == cv && oi < ci)) { cv = ov; ci = oi; }
      }
      if (t == 0) rowmax = cv;
      if (ci == lane)           k0 = true;
      else if (ci == lane + 32) k1 = true;
    }
    const float p0 = k0 ? __expf(s0 - rowmax) : 0.f;
    const float p1 = k1 ? __expf(s1 - rowmax) : 0.f;
    float sum = p0 + p1;
#pragma unroll
    for (int off = 16; off > 0; off >>= 1) sum += __shfl_xor(sum, off, 32);
    const float invs = 1.f / sum;
    r2.P[h][row][lane]      = (__bf16)(p0 * invs);
    r2.P[h][row][lane + 32] = (__bf16)(p1 * invs);
  }
  __syncthreads();  // all waves done with S -> r1 reusable as output staging

  // ---- Phase 5: out_head = P @ V -> staged in r1 for coalesced stores ------
#pragma unroll
  for (int mt = 0; mt < 4; ++mt) {
#pragma unroll
    for (int dt = 0; dt < 2; ++dt) {
      v8f acc = {};
#pragma unroll
      for (int kk = 0; kk < WA_N; kk += 32) {
        v16bf a  = frag_from_bf16(&r2.P[h][mt * 16][kk], WA_N, lane);
        v16bf vf = frag_from_bf16(&r3.vt[h][dt * 16][kk], WA_N, lane);
        acc = wmma_bf16(a, vf, acc);
      }
      const int dcol  = h * WA_HD + dt * 16 + (lane & 15);
      const int rbase = mt * 16 + ((lane >> 4) << 3);
#pragma unroll
      for (int v = 0; v < 8; ++v) {
        if (OB) r1.ostg_b[rbase + v][dcol] = (__bf16)acc[v];
        else    r1.ostg_f[rbase + v][dcol] = acc[v];
      }
    }
  }
  __syncthreads();

  // ---- Phase 6: coalesced tile store (16B per lane-store) ------------------
  if (OB) {
    v8bf* dst = (v8bf*)(aout_b + (size_t)b * WA_N * WA_C);
    const v8bf* src = (const v8bf*)&r1.ostg_b[0][0];
    for (int i = tid; i < WA_N * WA_C / 8; i += 256) dst[i] = src[i];
  } else {
    float4* dst = (float4*)(aout_f + (size_t)b * WA_N * WA_C);
    const float4* src = (const float4*)&r1.ostg_f[0][0];
    for (int i = tid; i < WA_N * WA_C / 4; i += 256) dst[i] = src[i];
  }
}

// ---------------------------------------------------------------------------
// Kernel B: y = a @ proj_w^T + proj_b.
// OB: input is bf16 in ws (separate buffer, fragments read straight from
//     global).  !OB: input aliases d_out -> stage 64 rows to LDS first
//     (output row n depends only on input row n, so in-place is safe).
// ---------------------------------------------------------------------------
template <bool WB, bool OB>
__global__ __launch_bounds__(256)
void wa_proj(const float*  ain_f,                    // [B,64,256] f32 (= d_out)
             const __bf16* __restrict__ ain_b,      // [B,64,256] bf16 in ws
             const float*  __restrict__ proj_w,     // [256,256] f32
             const __bf16* __restrict__ pwb,        // [256,256] bf16 (WB)
             const float*  __restrict__ proj_b,     // [256]
             float* out)                             // [B,64,256] (= d_out)
{
  __shared__ __align__(16) __bf16 tileA[WA_N][WA_C];  // 32KB (!OB staging)
  __shared__ __align__(16) float  ostg[WA_N][WA_C];   // 64KB coalescing stage
  const int b    = blockIdx.x;
  const int tid  = threadIdx.x;
  const int lane = tid & 31;
  const int wave = tid >> 5;

  const __bf16* abase;
  if (OB) {
    abase = ain_b + (size_t)b * WA_N * WA_C;
  } else {
    const float* ab = ain_f + (size_t)b * WA_N * WA_C;
    for (int e = tid; e < WA_N * WA_C; e += 256)
      tileA[e >> 8][e & 255] = (__bf16)ab[e];
    __syncthreads();
    abase = &tileA[0][0];
  }

  for (int t = wave; t < 4 * 16; t += 8) {
    const int mt = t & 3;
    const int ct = t >> 2;
    v8f acc = {};
#pragma unroll
    for (int kk = 0; kk < WA_C; kk += 32) {
      v16bf a  = frag_from_bf16(abase + (size_t)(mt * 16) * WA_C + kk, WA_C, lane);
      v16bf wf = WB ? frag_from_bf16(pwb + (size_t)(ct * 16) * WA_C + kk, WA_C, lane)
                    : frag_from_f32(proj_w + (size_t)(ct * 16) * WA_C + kk, WA_C, lane);
      acc = wmma_bf16(a, wf, acc);
    }
    const int   col   = ct * 16 + (lane & 15);
    const float bias  = proj_b[col];
    const int   rbase = mt * 16 + ((lane >> 4) << 3);
#pragma unroll
    for (int v = 0; v < 8; ++v)
      ostg[rbase + v][col] = acc[v] + bias;
  }
  __syncthreads();

  float4* dst = (float4*)(out + (size_t)b * WA_N * WA_C);
  const float4* src = (const float4*)&ostg[0][0];
  for (int i = tid; i < WA_N * WA_C / 4; i += 256) dst[i] = src[i];
}

extern "C" void kernel_launch(void* const* d_in, const int* in_sizes, int n_in,
                              void* d_out, int out_size, void* d_ws, size_t ws_size,
                              hipStream_t stream) {
  (void)n_in; (void)out_size;
  const float* x       = (const float*)d_in[0];
  const float* qkv_w   = (const float*)d_in[1];
  const float* qkv_b   = (const float*)d_in[2];
  const float* q_gamma = (const float*)d_in[3];
  const float* q_beta  = (const float*)d_in[4];
  const float* k_gamma = (const float*)d_in[5];
  const float* k_beta  = (const float*)d_in[6];
  const float* proj_w  = (const float*)d_in[7];
  const float* proj_b  = (const float*)d_in[8];
  float* out = (float*)d_out;

  const int B = in_sizes[0] / (WA_N * WA_C);  // 4096

  // Deterministic tier selection from ws_size (fixed across replays).
  const size_t needW = (size_t)(WA_QKVC * WA_C + WA_C * WA_C) * sizeof(__bf16);
  const size_t needA = (size_t)B * WA_N * WA_C * sizeof(__bf16);
  const bool wb = ws_size >= needW;
  const bool ob = ws_size >= needW + needA;

  __bf16* qwb   = (__bf16*)d_ws;
  __bf16* pwb   = qwb + WA_QKVC * WA_C;
  __bf16* aoutb = pwb + WA_C * WA_C;

  if (wb) wa_cvt_w<<<(WA_QKVC * WA_C + 255) / 256, 256, 0, stream>>>(qkv_w, proj_w,
                                                                     qwb, pwb);
  if (wb && ob) {
    wa_qkv_attn<true, true><<<B, 256, 0, stream>>>(x, qkv_w, qwb, qkv_b, q_gamma,
                                                   q_beta, k_gamma, k_beta, out, aoutb);
    wa_proj<true, true><<<B, 256, 0, stream>>>(out, aoutb, proj_w, pwb, proj_b, out);
  } else if (wb) {
    wa_qkv_attn<true, false><<<B, 256, 0, stream>>>(x, qkv_w, qwb, qkv_b, q_gamma,
                                                    q_beta, k_gamma, k_beta, out, nullptr);
    wa_proj<true, false><<<B, 256, 0, stream>>>(out, nullptr, proj_w, pwb, proj_b, out);
  } else {
    wa_qkv_attn<false, false><<<B, 256, 0, stream>>>(x, qkv_w, nullptr, qkv_b, q_gamma,
                                                     q_beta, k_gamma, k_beta, out, nullptr);
    wa_proj<false, false><<<B, 256, 0, stream>>>(out, nullptr, proj_w, nullptr, proj_b, out);
  }
}
